// Model_16595753632138
// MI455X (gfx1250) — matrix operand order repaired and checked by reading, not compiled, not run on hardware
//
#include <hip/hip_runtime.h>
#include <math.h>

#define SEQ_LEN 4096
#define DMODEL  2048
#define KC      32
#define LDA     40   // padded LDS stride in halves (80B, 16B-aligned rows)

typedef __attribute__((ext_vector_type(16))) _Float16 v16h;
typedef __attribute__((ext_vector_type(8)))  float    v8f;

union F16Frag { v16h v; uint4 q[2]; };

// ---- CDNA5 async global->LDS copy (ASYNCcnt-tracked, no VGPR round-trip) ----
__device__ __forceinline__ void async_ld128(unsigned lds_byte_off, const _Float16* g) {
  asm volatile("global_load_async_to_lds_b128 %0, %1, off"
               :: "v"(lds_byte_off), "v"(g) : "memory");
}
__device__ __forceinline__ void wait_async0() {
  asm volatile("s_wait_asynccnt 0x0" ::: "memory");
}

// ---------------- embedding gather: x (f32) + xh (f16) ----------------
__global__ __launch_bounds__(256) void embed_kernel(const int* __restrict__ ids,
    const float* __restrict__ emb, float* __restrict__ x, _Float16* __restrict__ xh) {
  int row = blockIdx.x;
  int id  = ids[row];
  const float* e = emb + (size_t)id * DMODEL;
  float*     xo  = x  + (size_t)row * DMODEL;
  _Float16*  xho = xh + (size_t)row * DMODEL;
  for (int j = threadIdx.x; j < DMODEL; j += blockDim.x) {
    float v = e[j];
    xo[j]  = v;
    xho[j] = (_Float16)v;
  }
}

// ---------------- weight convert + transpose: W[k][n] f32 -> Wt[n][k] f16 ----------------
__global__ __launch_bounds__(256) void wcvt_kernel(const float* __restrict__ W,
    _Float16* __restrict__ Wt, int Kd, int Nd) {
  __shared__ float tile[32][33];
  int kb = blockIdx.x * 32;
  int nb = blockIdx.y * 32;
  int tx = threadIdx.x & 31;
  int ty = threadIdx.x >> 5;                 // 0..7
  for (int i = ty; i < 32; i += 8)
    tile[i][tx] = W[(size_t)(kb + i) * Nd + (nb + tx)];
  __syncthreads();
  for (int i = ty; i < 32; i += 8)
    Wt[(size_t)(nb + i) * Kd + (kb + tx)] = (_Float16)tile[tx][i];
}

// ---------------- unified NT WMMA GEMM, 128x64 tile, async double-buffered ----------------
// C[M,N] = A[M,K] * B[N,K]^T   (A,B f16, K-contiguous rows)
// EPI 0: C f16 row-major, + bias[n]            (Q, K projections)
// EPI 1: C f16 stored transposed [N][M], +bias (V projection -> vt)
// EPI 2: C f32 row-major, causal mask; skip tiles above diagonal
// EPI 3: C f32 = resid + acc; also Caux f16 shadow (attn@V + residual)
template <int EPI>
__global__ __launch_bounds__(128) void gemm_nt(
    const _Float16* __restrict__ A, const _Float16* __restrict__ B,
    const float* __restrict__ bias, const float* __restrict__ resid,
    void* __restrict__ Cmain, _Float16* __restrict__ Caux,
    int M, int N, int K) {
  const int m0 = blockIdx.y * 128;
  const int n0 = blockIdx.x * 64;
  if (EPI == 2 && n0 >= m0 + 128) return;    // whole tile above causal diagonal

  __shared__ _Float16 Alds[2][128 * LDA];    // 2 x 10 KB
  __shared__ _Float16 Blds[2][64 * LDA];     // 2 x  5 KB

  const int tid  = threadIdx.x;
  const int wid  = tid >> 5;                 // wave 0..3 -> 32-row M strip
  const int lane = tid & 31;

  // per-thread transfer assignment: one 32-half A row, half a B row
  const _Float16* ga = A + (size_t)(m0 + tid) * K;
  const int brow = tid >> 1, bcol = (tid & 1) << 4;
  const _Float16* gb = B + (size_t)(n0 + brow) * K + bcol;

  // LDS byte addresses for this thread's transfer targets (low 32 bits of
  // the generic pointer == DS offset per the flat->LDS aperture rule)
  const unsigned aL0 = (unsigned)(uintptr_t)&Alds[0][tid * LDA];
  const unsigned aL1 = (unsigned)(uintptr_t)&Alds[1][tid * LDA];
  const unsigned bL0 = (unsigned)(uintptr_t)&Blds[0][brow * LDA + bcol];
  const unsigned bL1 = (unsigned)(uintptr_t)&Blds[1][brow * LDA + bcol];

  v8f acc[2][4];
  #pragma unroll
  for (int s = 0; s < 2; ++s)
    #pragma unroll
    for (int t = 0; t < 4; ++t)
      #pragma unroll
      for (int r = 0; r < 8; ++r) acc[s][t][r] = 0.0f;

  // fragment offsets (in halves) within one LDS buffer
  const int aoff0 = (wid * 32 + (lane & 15)) * LDA + ((lane >> 4) << 4);
  const int aoff1 = aoff0 + 16 * LDA;
  const int boff  = (lane & 15) * LDA + ((lane >> 4) << 4);

  // issue tile 0 into buffer 0
  {
    async_ld128(aL0,      ga);
    async_ld128(aL0 + 16, ga + 8);
    async_ld128(aL0 + 32, ga + 16);
    async_ld128(aL0 + 48, ga + 24);
    async_ld128(bL0,      gb);
    async_ld128(bL0 + 16, gb + 8);
  }

  int buf = 0;
  for (int k0 = 0; k0 < K; k0 += KC) {
    wait_async0();                  // this buffer's 6 transfers complete
    __syncthreads();                // ... and visible to all waves

    if (k0 + KC < K) {              // kick off next tile into other buffer
      const _Float16* gan = ga + k0 + KC;
      const _Float16* gbn = gb + k0 + KC;
      const unsigned aLn = buf ? aL0 : aL1;
      const unsigned bLn = buf ? bL0 : bL1;
      async_ld128(aLn,      gan);
      async_ld128(aLn + 16, gan + 8);
      async_ld128(aLn + 32, gan + 16);
      async_ld128(aLn + 48, gan + 24);
      async_ld128(bLn,      gbn);
      async_ld128(bLn + 16, gbn + 8);
      __builtin_prefetch(gan + KC, 0, 1);
      __builtin_prefetch(gbn + KC, 0, 1);
    }

    // load ALL fragments, then 8 WMMAs back-to-back
    const _Float16* lbuf = &Alds[buf][0];
    F16Frag af0, af1, bf0, bf1, bf2, bf3;
    af0.q[0] = *(const uint4*)(lbuf + aoff0);
    af0.q[1] = *(const uint4*)(lbuf + aoff0 + 8);
    af1.q[0] = *(const uint4*)(lbuf + aoff1);
    af1.q[1] = *(const uint4*)(lbuf + aoff1 + 8);
    const _Float16* bbuf = &Blds[buf][0];
    bf0.q[0] = *(const uint4*)(bbuf + boff);
    bf0.q[1] = *(const uint4*)(bbuf + boff + 8);
    bf1.q[0] = *(const uint4*)(bbuf + 16 * LDA + boff);
    bf1.q[1] = *(const uint4*)(bbuf + 16 * LDA + boff + 8);
    bf2.q[0] = *(const uint4*)(bbuf + 32 * LDA + boff);
    bf2.q[1] = *(const uint4*)(bbuf + 32 * LDA + boff + 8);
    bf3.q[0] = *(const uint4*)(bbuf + 48 * LDA + boff);
    bf3.q[1] = *(const uint4*)(bbuf + 48 * LDA + boff + 8);

    acc[0][0] = __builtin_amdgcn_wmma_f32_16x16x32_f16(false, af0.v, false, bf0.v, (short)0, acc[0][0], false, false);
    acc[1][0] = __builtin_amdgcn_wmma_f32_16x16x32_f16(false, af1.v, false, bf0.v, (short)0, acc[1][0], false, false);
    acc[0][1] = __builtin_amdgcn_wmma_f32_16x16x32_f16(false, af0.v, false, bf1.v, (short)0, acc[0][1], false, false);
    acc[1][1] = __builtin_amdgcn_wmma_f32_16x16x32_f16(false, af1.v, false, bf1.v, (short)0, acc[1][1], false, false);
    acc[0][2] = __builtin_amdgcn_wmma_f32_16x16x32_f16(false, af0.v, false, bf2.v, (short)0, acc[0][2], false, false);
    acc[1][2] = __builtin_amdgcn_wmma_f32_16x16x32_f16(false, af1.v, false, bf2.v, (short)0, acc[1][2], false, false);
    acc[0][3] = __builtin_amdgcn_wmma_f32_16x16x32_f16(false, af0.v, false, bf3.v, (short)0, acc[0][3], false, false);
    acc[1][3] = __builtin_amdgcn_wmma_f32_16x16x32_f16(false, af1.v, false, bf3.v, (short)0, acc[1][3], false, false);

    buf ^= 1;
  }

  // C layout: vgpr r, lanes 0-15 -> M=r, N=lane; lanes 16-31 -> M=r+8, N=lane-16
  const int ncol = lane & 15;

  #pragma unroll
  for (int s = 0; s < 2; ++s) {
    const int mbase = m0 + wid * 32 + s * 16 + ((lane >> 4) << 3);
    if (EPI == 0) {
      _Float16* C = (_Float16*)Cmain;
      #pragma unroll
      for (int t = 0; t < 4; ++t) {
        int n = n0 + t * 16 + ncol;
        float b = bias[n];
        #pragma unroll
        for (int r = 0; r < 8; ++r)
          C[(size_t)(mbase + r) * N + n] = (_Float16)(acc[s][t][r] + b);
      }
    } else if (EPI == 1) {
      _Float16* C = (_Float16*)Cmain;        // [N][M]
      #pragma unroll
      for (int t = 0; t < 4; ++t) {
        int n = n0 + t * 16 + ncol;
        float b = bias[n];
        #pragma unroll
        for (int r = 0; r < 8; ++r)
          C[(size_t)n * M + (mbase + r)] = (_Float16)(acc[s][t][r] + b);
      }
    } else if (EPI == 2) {
      float* C = (float*)Cmain;
      #pragma unroll
      for (int t = 0; t < 4; ++t) {
        int n = n0 + t * 16 + ncol;
        #pragma unroll
        for (int r = 0; r < 8; ++r) {
          int m = mbase + r;
          C[(size_t)m * N + n] = (n <= m) ? acc[s][t][r] : -__builtin_inff();
        }
      }
    } else {
      float* C = (float*)Cmain;
      #pragma unroll
      for (int t = 0; t < 4; ++t) {
        int n = n0 + t * 16 + ncol;
        #pragma unroll
        for (int r = 0; r < 8; ++r) {
          size_t idx = (size_t)(mbase + r) * N + n;
          float v = resid[idx] + acc[s][t][r];
          C[idx]    = v;
          Caux[idx] = (_Float16)v;
        }
      }
    }
  }
}

// ---------------- causal row softmax: S f32 -> P f16 ----------------
__global__ __launch_bounds__(256) void softmax_kernel(const float* __restrict__ S,
    _Float16* __restrict__ P) {
  int row   = blockIdx.x;
  int valid = row + 1;
  const float* s = S + (size_t)row * SEQ_LEN;
  _Float16*    p = P + (size_t)row * SEQ_LEN;
  __shared__ float red[256];
  int tid = threadIdx.x;

  float mx = -3.4e38f;
  for (int j = tid; j < valid; j += 256) mx = fmaxf(mx, s[j]);
  red[tid] = mx; __syncthreads();
  for (int st = 128; st > 0; st >>= 1) {
    if (tid < st) red[tid] = fmaxf(red[tid], red[tid + st]);
    __syncthreads();
  }
  mx = red[0]; __syncthreads();

  float sum = 0.0f;
  for (int j = tid; j < valid; j += 256) sum += __expf(s[j] - mx);
  red[tid] = sum; __syncthreads();
  for (int st = 128; st > 0; st >>= 1) {
    if (tid < st) red[tid] += red[tid + st];
    __syncthreads();
  }
  float inv = 1.0f / red[0];

  for (int j = tid; j < SEQ_LEN; j += 256)
    p[j] = (j < valid) ? (_Float16)(__expf(s[j] - mx) * inv) : (_Float16)0.0f;
}

// ---------------- host launcher ----------------
extern "C" void kernel_launch(void* const* d_in, const int* in_sizes, int n_in,
                              void* d_out, int out_size, void* d_ws, size_t ws_size,
                              hipStream_t stream) {
  (void)in_sizes; (void)n_in; (void)out_size; (void)ws_size;

  const int*   ids = (const int*)d_in[0];
  const float* emb = (const float*)d_in[1];
  // order: Wq1 bq1 Wk1 bk1 Wv1 bv1 Wq2 bq2 Wk2 bk2 Wv2 bv2
  const float* W[6]  = { (const float*)d_in[2],  (const float*)d_in[4],  (const float*)d_in[6],
                         (const float*)d_in[8],  (const float*)d_in[10], (const float*)d_in[12] };
  const float* Bv[6] = { (const float*)d_in[3],  (const float*)d_in[5],  (const float*)d_in[7],
                         (const float*)d_in[9],  (const float*)d_in[11], (const float*)d_in[13] };

  char* ws = (char*)d_ws;
  size_t off = 0;
  auto carve = [&](size_t bytes) -> void* {
    void* p = ws + off;
    off += (bytes + 255) & ~(size_t)255;
    return p;
  };
  float*    x  = (float*)   carve((size_t)SEQ_LEN * DMODEL * 4);
  _Float16* xh = (_Float16*)carve((size_t)SEQ_LEN * DMODEL * 2);
  _Float16* qh = (_Float16*)carve((size_t)SEQ_LEN * DMODEL * 2);
  _Float16* kh = (_Float16*)carve((size_t)SEQ_LEN * DMODEL * 2);
  _Float16* vt = (_Float16*)carve((size_t)DMODEL * SEQ_LEN * 2);   // [D][S]
  float*    Sc = (float*)   carve((size_t)SEQ_LEN * SEQ_LEN * 4);  // L2-resident
  _Float16* Pm = (_Float16*)carve((size_t)SEQ_LEN * SEQ_LEN * 2);  // L2-resident
  _Float16* Wt[6];
  for (int i = 0; i < 6; ++i)
    Wt[i] = (_Float16*)carve((size_t)DMODEL * DMODEL * 2);

  embed_kernel<<<SEQ_LEN, 256, 0, stream>>>(ids, emb, x, xh);

  dim3 cgrid(DMODEL / 32, DMODEL / 32);
  for (int i = 0; i < 6; ++i)
    wcvt_kernel<<<cgrid, 256, 0, stream>>>(W[i], Wt[i], DMODEL, DMODEL);

  for (int l = 0; l < 2; ++l) {
    dim3 gProj(DMODEL / 64, SEQ_LEN / 128);
    gemm_nt<0><<<gProj, 128, 0, stream>>>(xh, Wt[3*l+0], Bv[3*l+0], nullptr,
                                          qh, nullptr, SEQ_LEN, DMODEL, DMODEL);
    gemm_nt<0><<<gProj, 128, 0, stream>>>(xh, Wt[3*l+1], Bv[3*l+1], nullptr,
                                          kh, nullptr, SEQ_LEN, DMODEL, DMODEL);
    gemm_nt<1><<<gProj, 128, 0, stream>>>(xh, Wt[3*l+2], Bv[3*l+2], nullptr,
                                          vt, nullptr, SEQ_LEN, DMODEL, DMODEL);

    dim3 gScore(SEQ_LEN / 64, SEQ_LEN / 128);
    gemm_nt<2><<<gScore, 128, 0, stream>>>(qh, kh, nullptr, nullptr,
                                           Sc, nullptr, SEQ_LEN, SEQ_LEN, DMODEL);

    softmax_kernel<<<SEQ_LEN, 256, 0, stream>>>(Sc, Pm);

    float* outp = (l == 1) ? (float*)d_out : x;
    gemm_nt<3><<<gProj, 128, 0, stream>>>(Pm, vt, nullptr, x,
                                          outp, xh, SEQ_LEN, DMODEL, SEQ_LEN);
  }
}
